// RBFRNN_89421219103092
// MI455X (gfx1250) — matrix operand order repaired and checked by reading, not compiled, not run on hardware
//
#include <hip/hip_runtime.h>
#include <hip/hip_bf16.h>

typedef __attribute__((ext_vector_type(16))) __bf16 bf16x16;
typedef __attribute__((ext_vector_type(8)))  float  f32x8;

#define NODES   1024
#define ONODES  512
#define BB      64
#define TT      256
#define RECBLK  32      // persistent blocks for the recurrence (32 cols each)

__device__ __forceinline__ unsigned short f2bf(float f) {
  unsigned u = __float_as_uint(f);
  u += 0x7FFFu + ((u >> 16) & 1u);
  return (unsigned short)(u >> 16);
}

union BF16Frag {
  struct { uint4 lo, hi; } q;
  bf16x16 v;
};

// ---------------------------------------------------------------------------
// Pack a row-major f32 K x N matrix into per-lane WMMA-B bf16 fragments (contiguous k order, matching A):
// dst[((nt*(K/32)+kt)*32 + lane)*16 + j] = src[(kt*32 + (lane>=16?16:0) + j)*N + nt*16 + (lane&15)]
// ---------------------------------------------------------------------------
__global__ void pack_b_kernel(const float* __restrict__ src,
                              unsigned short* __restrict__ dst,
                              int K, int N) {
  int idx = blockIdx.x * blockDim.x + threadIdx.x;
  if (idx >= K * N) return;
  int j    = idx & 15;
  int lane = (idx >> 4) & 31;
  int tile = idx >> 9;
  int Kt   = K >> 5;
  int kt   = tile % Kt;
  int nt   = tile / Kt;
  int k = kt * 32 + ((lane >> 4) << 4) + j;
  int n = nt * 16 + (lane & 15);
  dst[idx] = f2bf(src[(size_t)k * N + n]);
}

// h0 = broadcast(init_state) bf16 (64 x 1024); also zero the grid-barrier counter
__global__ void init_h0_kernel(const float* __restrict__ init_state,
                               unsigned short* __restrict__ h0,
                               unsigned int* __restrict__ counter) {
  int idx = blockIdx.x * blockDim.x + threadIdx.x;   // 64*1024
  h0[idx] = f2bf(init_state[idx & (NODES - 1)]);
  if (idx == 0) *counter = 0u;
}

// ---------------------------------------------------------------------------
// drive[t][b][n] = sum_k sf[k]*speed_W[k][n] + sum_k hf[k]*hd_W[k][n]
// ---------------------------------------------------------------------------
__global__ void drive_kernel(const float* __restrict__ x,
                             const float* __restrict__ speed_W,
                             const float* __restrict__ hd_W,
                             float* __restrict__ drive) {
  __shared__ float sf[20];
  __shared__ float hf[50];
  int bid = blockIdx.x;            // t*64 + b
  int t = bid >> 6;
  int b = bid & 63;
  int tid = threadIdx.x;
  float v  = x[((size_t)b * TT + t) * 2 + 0];
  float hd = x[((size_t)b * TT + t) * 2 + 1];
  if (tid < 20) {
    float c = 0.5f * (float)tid / 19.0f;
    float d = v - c;
    sf[tid] = __expf(-800.0f * d * d);          // 0.5/sd^2, sd = 0.025
  }
  if (tid < 50) {
    float c = 6.28318530717958647f * (float)tid / 49.0f;
    hf[tid] = __expf(6.28318530717958647f * __cosf(hd - c) - 6.305047f);
  }
  __syncthreads();
  float acc[4] = {0.f, 0.f, 0.f, 0.f};
  for (int k = 0; k < 20; ++k) {
    float s = sf[k];
    const float* wrow = speed_W + (size_t)k * NODES;
#pragma unroll
    for (int j = 0; j < 4; ++j) acc[j] += s * wrow[tid + 256 * j];
  }
  for (int k = 0; k < 50; ++k) {
    float s = hf[k];
    const float* wrow = hd_W + (size_t)k * NODES;
#pragma unroll
    for (int j = 0; j < 4; ++j) acc[j] += s * wrow[tid + 256 * j];
  }
  float* drow = drive + (size_t)bid * NODES;
#pragma unroll
  for (int j = 0; j < 4; ++j) drow[tid + 256 * j] = acc[j];
}

// ---------------------------------------------------------------------------
// Persistent recurrence: all 256 steps in one kernel.
// grid = 32 blocks; block b owns cols [b*32, b*32+32) for every step.
// Its 64 KB slice of WMMA-packed rec_W is async-copied to LDS once and
// reused 256 times; steps are separated by an L2 grid barrier.
// ---------------------------------------------------------------------------
__global__ void __launch_bounds__(256)
rec_persistent_kernel(unsigned short* __restrict__ hsext,
                      const unsigned short* __restrict__ recWp,
                      const float* __restrict__ drive,
                      const float* __restrict__ rec_b,
                      unsigned int* __restrict__ counter) {
  __shared__ uint4 ldsB4[4096];                       // 64 KB, 16B aligned
  unsigned short* ldsB = (unsigned short*)ldsB4;

  const int tid     = threadIdx.x;
  const int lane    = tid & 31;
  const int w       = tid >> 5;
  const int mtile   = w & 3;                          // 4 M-tiles of 16 rows
  const int ntLocal = w >> 2;                         // 2 local N-tiles
  const int laneHi  = lane >> 4;
  const int laneLo  = lane & 15;
  const int ntGlobal = blockIdx.x * 2 + ntLocal;

  // --- stage this block's packed rec_W slice (contiguous 64 KB) into LDS ---
  {
    const unsigned short* src = recWp + (size_t)blockIdx.x * 2 * 32 * 512;
    unsigned ldsBase = (unsigned)(size_t)(void*)ldsB;
#pragma unroll 4
    for (int i = 0; i < 16; ++i) {
      unsigned off = (unsigned)(i * 256 + tid) * 16u;
      unsigned long long gaddr = (unsigned long long)(size_t)src + off;
      asm volatile("global_load_async_to_lds_b128 %0, %1, off"
                   :: "v"(ldsBase + off), "v"(gaddr)
                   : "memory");
    }
    asm volatile("s_wait_asynccnt 0x0" ::: "memory");
  }
  __syncthreads();

  const float bias = rec_b[ntGlobal * 16 + laneLo];
  const unsigned short* bbase =
      ldsB + ((size_t)ntLocal * 32 * 32 + lane) * 16; // + kt*512 per K-tile

  for (int t = 0; t < TT; ++t) {
    const unsigned short* hprev = hsext + (size_t)t * BB * NODES;
    unsigned short* hnext       = hsext + (size_t)(t + 1) * BB * NODES;
    const float* drive_t        = drive + (size_t)t * BB * NODES;

    f32x8 acc = {};
    const unsigned short* arow =
        hprev + (size_t)(mtile * 16 + laneLo) * NODES;

    for (int kt = 0; kt < 32; ++kt) {
      int koff = kt * 32 + laneHi * 16;
      BF16Frag a, b;
      a.q.lo = *(const uint4*)(arow + koff);
      a.q.hi = *(const uint4*)(arow + koff + 8);
      const unsigned short* bp = bbase + kt * 512;
      b.q.lo = *(const uint4*)(bp);
      b.q.hi = *(const uint4*)(bp + 8);
      acc = __builtin_amdgcn_wmma_f32_16x16x32_bf16(false, a.v, false, b.v,
                                                    (short)0, acc, false, false);
    }

    int rbase = mtile * 16 + laneHi * 8;
    int c = ntGlobal * 16 + laneLo;
#pragma unroll
    for (int i = 0; i < 8; ++i) {
      int rr = rbase + i;
      float v = acc[i] + bias + drive_t[(size_t)rr * NODES + c];
      v = v > 0.f ? v : 0.f;
      hnext[(size_t)rr * NODES + c] = f2bf(v);
    }

    // ---- grid barrier: monotonic counter, release/acquire at agent scope ----
    __threadfence();
    __syncthreads();
    if (tid == 0) {
      __hip_atomic_fetch_add(counter, 1u, __ATOMIC_RELEASE,
                             __HIP_MEMORY_SCOPE_AGENT);
      unsigned target = (unsigned)(RECBLK) * (unsigned)(t + 1);
      while (__hip_atomic_load(counter, __ATOMIC_ACQUIRE,
                               __HIP_MEMORY_SCOPE_AGENT) < target) {
        __builtin_amdgcn_s_sleep(1);
      }
      __threadfence();
    }
    __syncthreads();
  }
}

// ---------------------------------------------------------------------------
// out = relu(hs @ out_W + out_b); hs = (T*B, 1024) bf16, out = (T*B, 512) f32
// grid = (256, 8): 64 rows x 64 cols per block
// ---------------------------------------------------------------------------
__global__ void __launch_bounds__(256)
out_gemm_kernel(const unsigned short* __restrict__ hs,
                const unsigned short* __restrict__ outWp,
                const float* __restrict__ out_b,
                float* __restrict__ outbuf) {
  int lane   = threadIdx.x & 31;
  int w      = threadIdx.x >> 5;
  int mtile  = w & 3;
  int laneHi = lane >> 4;
  int laneLo = lane & 15;
  int rowBase = blockIdx.x * 64;
  int nt0 = blockIdx.y * 4 + (w >> 2) * 2;
  int nt1 = nt0 + 1;

  f32x8 acc0 = {};
  f32x8 acc1 = {};
  const unsigned short* arow = hs + (size_t)(rowBase + mtile * 16 + laneLo) * NODES;

  for (int kt = 0; kt < 32; ++kt) {
    int koff = kt * 32 + laneHi * 16;
    BF16Frag a, b0, b1;
    a.q.lo = *(const uint4*)(arow + koff);
    a.q.hi = *(const uint4*)(arow + koff + 8);
    const unsigned short* bp0 = outWp + (((size_t)nt0 * 32 + kt) * 32 + lane) * 16;
    const unsigned short* bp1 = outWp + (((size_t)nt1 * 32 + kt) * 32 + lane) * 16;
    b0.q.lo = *(const uint4*)(bp0);
    b0.q.hi = *(const uint4*)(bp0 + 8);
    b1.q.lo = *(const uint4*)(bp1);
    b1.q.hi = *(const uint4*)(bp1 + 8);
    acc0 = __builtin_amdgcn_wmma_f32_16x16x32_bf16(false, a.v, false, b0.v,
                                                   (short)0, acc0, false, false);
    acc1 = __builtin_amdgcn_wmma_f32_16x16x32_bf16(false, a.v, false, b1.v,
                                                   (short)0, acc1, false, false);
  }

  int rbase = rowBase + mtile * 16 + laneHi * 8;
  int c0 = nt0 * 16 + laneLo;
  int c1 = nt1 * 16 + laneLo;
  float bias0 = out_b[c0], bias1 = out_b[c1];
#pragma unroll
  for (int i = 0; i < 8; ++i) {
    int r = rbase + i;
    float v0 = acc0[i] + bias0;
    float v1 = acc1[i] + bias1;
    v0 = v0 > 0.f ? v0 : 0.f;
    v1 = v1 > 0.f ? v1 : 0.f;
    outbuf[(size_t)r * ONODES + c0] = v0;
    outbuf[(size_t)r * ONODES + c1] = v1;
  }
}

// ---------------------------------------------------------------------------
// centers[b][n][c] = sum_t softmax_t(out[b,:,n])[t] * r[b,t,c]
// one wave per (b,n); out stored as (t*64+b)*512 + n
// ---------------------------------------------------------------------------
__global__ void centers_kernel(const float* __restrict__ outbuf,
                               const float* __restrict__ rbuf,
                               float* __restrict__ centers) {
  int lane = threadIdx.x & 31;
  int w    = threadIdx.x >> 5;
  int idx  = blockIdx.x * 8 + w;    // b*512 + n
  int b = idx >> 9;
  int n = idx & 511;

  float vals[8];
  float m = -1e30f;
#pragma unroll
  for (int j = 0; j < 8; ++j) {
    int t = lane + j * 32;
    vals[j] = outbuf[((size_t)t * 64 + b) * ONODES + n];
    m = fmaxf(m, vals[j]);
  }
  for (int off = 16; off; off >>= 1) m = fmaxf(m, __shfl_xor(m, off, 32));

  float se = 0.f, s0 = 0.f, s1 = 0.f;
#pragma unroll
  for (int j = 0; j < 8; ++j) {
    int t = lane + j * 32;
    float e = __expf(vals[j] - m);
    se += e;
    s0 += e * rbuf[((size_t)b * TT + t) * 2 + 0];
    s1 += e * rbuf[((size_t)b * TT + t) * 2 + 1];
  }
  for (int off = 16; off; off >>= 1) {
    se += __shfl_xor(se, off, 32);
    s0 += __shfl_xor(s0, off, 32);
    s1 += __shfl_xor(s1, off, 32);
  }
  if (lane == 0) {
    float inv = 1.f / se;
    centers[(size_t)idx * 2 + 0] = s0 * inv;
    centers[(size_t)idx * 2 + 1] = s1 * inv;
  }
}

// ---------------------------------------------------------------------------
// final[b][t][c] = sum_n softmax_n(out[b,t,:])[n] * centers[b,n,c]
// ---------------------------------------------------------------------------
__global__ void final_kernel(const float* __restrict__ outbuf,
                             const float* __restrict__ centers,
                             float* __restrict__ out) {
  __shared__ float red[256];
  int bid = blockIdx.x;            // b*256 + t
  int b = bid >> 8;
  int t = bid & 255;
  int tid = threadIdx.x;

  const float* orow = outbuf + ((size_t)t * 64 + b) * ONODES;
  float v0 = orow[tid], v1 = orow[tid + 256];

  float m = fmaxf(v0, v1);
  red[tid] = m; __syncthreads();
  for (int s = 128; s; s >>= 1) { if (tid < s) red[tid] = fmaxf(red[tid], red[tid + s]); __syncthreads(); }
  m = red[0]; __syncthreads();

  float e0 = __expf(v0 - m), e1 = __expf(v1 - m);
  red[tid] = e0 + e1; __syncthreads();
  for (int s = 128; s; s >>= 1) { if (tid < s) red[tid] += red[tid + s]; __syncthreads(); }
  float inv = 1.f / red[0]; __syncthreads();

  const float* cb = centers + (size_t)b * ONODES * 2;
  float a0 = e0 * cb[tid * 2 + 0] + e1 * cb[(tid + 256) * 2 + 0];
  float a1 = e0 * cb[tid * 2 + 1] + e1 * cb[(tid + 256) * 2 + 1];

  red[tid] = a0; __syncthreads();
  for (int s = 128; s; s >>= 1) { if (tid < s) red[tid] += red[tid + s]; __syncthreads(); }
  float s0 = red[0]; __syncthreads();

  red[tid] = a1; __syncthreads();
  for (int s = 128; s; s >>= 1) { if (tid < s) red[tid] += red[tid + s]; __syncthreads(); }
  float s1 = red[0];

  if (tid == 0) {
    out[(size_t)bid * 2 + 0] = s0 * inv;
    out[(size_t)bid * 2 + 1] = s1 * inv;
  }
}

extern "C" void kernel_launch(void* const* d_in, const int* in_sizes, int n_in,
                              void* d_out, int out_size, void* d_ws, size_t ws_size,
                              hipStream_t stream) {
  const float* x          = (const float*)d_in[0];
  const float* r          = (const float*)d_in[1];
  const float* speed_W    = (const float*)d_in[2];
  const float* hd_W       = (const float*)d_in[3];
  const float* rec_W      = (const float*)d_in[4];
  const float* rec_b      = (const float*)d_in[5];
  const float* out_W      = (const float*)d_in[6];
  const float* out_b      = (const float*)d_in[7];
  const float* init_state = (const float*)d_in[8];

  char* ws = (char*)d_ws;
  size_t off = 0;
  auto alloc = [&](size_t bytes) {
    void* p = ws + off;
    off = (off + bytes + 255) & ~(size_t)255;
    return p;
  };
  unsigned short* recWp = (unsigned short*)alloc((size_t)NODES * NODES * 2);
  unsigned short* outWp = (unsigned short*)alloc((size_t)NODES * ONODES * 2);
  unsigned short* hsext = (unsigned short*)alloc((size_t)(TT + 1) * BB * NODES * 2);
  float* drive          = (float*)alloc((size_t)TT * BB * NODES * 4);
  float* outbuf         = (float*)alloc((size_t)TT * BB * ONODES * 4);
  float* centers        = (float*)alloc((size_t)BB * ONODES * 2 * 4);
  unsigned int* counter = (unsigned int*)alloc(256);

  pack_b_kernel<<<(NODES * NODES + 255) / 256, 256, 0, stream>>>(rec_W, recWp, NODES, NODES);
  pack_b_kernel<<<(NODES * ONODES + 255) / 256, 256, 0, stream>>>(out_W, outWp, NODES, ONODES);
  init_h0_kernel<<<(BB * NODES) / 256, 256, 0, stream>>>(init_state, hsext, counter);
  drive_kernel<<<TT * BB, 256, 0, stream>>>(x, speed_W, hd_W, drive);

  rec_persistent_kernel<<<RECBLK, 256, 0, stream>>>(hsext, recWp, drive, rec_b, counter);

  out_gemm_kernel<<<dim3(TT * BB / 64, ONODES / 64), 256, 0, stream>>>(
      hsext + (size_t)BB * NODES, outWp, out_b, outbuf);

  centers_kernel<<<BB * ONODES / 8, 256, 0, stream>>>(outbuf, r, centers);
  final_kernel<<<BB * TT, 256, 0, stream>>>(outbuf, centers, (float*)d_out);
}
